// TrajGPT_31851477467508
// MI455X (gfx1250) — matrix operand order repaired and checked by reading, not compiled, not run on hardware
//
#include <hip/hip_runtime.h>

#define DEV __device__ __forceinline__

typedef __attribute__((ext_vector_type(16))) _Float16 v16h;
typedef __attribute__((ext_vector_type(8)))  _Float16 v8h;
typedef __attribute__((ext_vector_type(8)))  float    v8f;
typedef __attribute__((ext_vector_type(4)))  int      v4i;

constexpr int LAYERS = 2, D = 1024, H = 16, QKD = 1024, VDIM = 1024, FDIM = 4096, VOCAB = 2000;
constexpr int B = 4, T = 1024;
constexpr int M_TOK = B * T;            // 4096 tokens
constexpr int QKV_N = 2 * QKD + VDIM;   // 3072
constexpr int QKV_STRIDE = QKV_N;

// ---------------------------------------------------------------------------
// Fragment layouts (wave32, CDNA5 ISA 7.12.2):
//  A 16x32 f16 : lane%16 = M row; half=lane/16.
//                elements 0-7  = K  (half?8:0)+0..7
//                elements 8-15 = K 16+(half?8:0)+0..7   (contiguous 8-chunks)
//  B 32x16 f16 : lane%16 = N column; half=lane/16, element s = K row
//                -> K order per element same as A (table order).
//  C/D 16x16 f32: VGPR j -> row j + 8*(lane>>4), col = lane&15.
// ---------------------------------------------------------------------------
DEV v16h cat8(v8h lo, v8h hi) {
  return __builtin_shufflevector(lo, hi, 0, 1, 2, 3, 4, 5, 6, 7,
                                 8, 9, 10, 11, 12, 13, 14, 15);
}

DEV v8f wmma_f16(v16h a, v16h b, v8f c) {
  return __builtin_amdgcn_wmma_f32_16x16x32_f16(
      /*neg_a=*/false, a, /*neg_b=*/false, b,
      /*c_mod=*/(short)0, c, /*reuse_a=*/false, /*reuse_b=*/false);
}

// ---- CDNA5 async global->LDS copy path ---------
#if defined(__HIP_DEVICE_COMPILE__) && __has_builtin(__builtin_amdgcn_global_load_async_to_lds_b128)
#define USE_ASYNC_COPY 1
#else
#define USE_ASYNC_COPY 0
#endif

DEV void async_wait0() {
#if defined(__HIP_DEVICE_COMPILE__)
#if __has_builtin(__builtin_amdgcn_s_wait_asynccnt)
  __builtin_amdgcn_s_wait_asynccnt(0);
#else
  asm volatile("s_wait_asynccnt 0" ::: "memory");
#endif
#endif
}

struct alignas(16) H8 { _Float16 v[8]; };

// ---------------------------------------------------------------------------
// Embedding + shift-right with SOS:  h[b,t,:] = t==0 ? sos : embed[X[b,t-1]]
// ---------------------------------------------------------------------------
__global__ __launch_bounds__(256) void embed_kernel(const int* __restrict__ X,
                                                    const float* __restrict__ embed,
                                                    const float* __restrict__ sos,
                                                    float* __restrict__ h) {
  int tok = blockIdx.x;
  int b = tok / T, t = tok % T;
  const float* src = (t == 0) ? sos : (embed + (size_t)X[b * T + t - 1] * D);
  float* dst = h + (size_t)tok * D;
  for (int i = threadIdx.x; i < D; i += 256) dst[i] = src[i];
}

// ---------------------------------------------------------------------------
// LayerNorm over D=1024 per token -> f16 output (GEMM input precision)
// ---------------------------------------------------------------------------
__global__ __launch_bounds__(256) void ln_kernel(const float* __restrict__ x,
                                                 const float* __restrict__ sc,
                                                 const float* __restrict__ bs,
                                                 _Float16* __restrict__ y) {
  int tok = blockIdx.x;
  const float* xp = x + (size_t)tok * D;
  float s0 = 0.f, s1 = 0.f;
  for (int i = threadIdx.x; i < D; i += 256) { float v = xp[i]; s0 += v; s1 += v * v; }
  for (int o = 16; o > 0; o >>= 1) { s0 += __shfl_down(s0, o, 32); s1 += __shfl_down(s1, o, 32); }
  __shared__ float r0[8], r1[8], mv[2];
  int lane = threadIdx.x & 31, wv = threadIdx.x >> 5;
  if (lane == 0) { r0[wv] = s0; r1[wv] = s1; }
  __syncthreads();
  if (threadIdx.x == 0) {
    float a = 0.f, q = 0.f;
    for (int i = 0; i < 8; ++i) { a += r0[i]; q += r1[i]; }
    float mean = a / D;
    mv[0] = mean;
    mv[1] = rsqrtf(q / D - mean * mean + 1e-5f);
  }
  __syncthreads();
  float mean = mv[0], rs = mv[1];
  _Float16* yp = y + (size_t)tok * D;
  for (int i = threadIdx.x; i < D; i += 256)
    yp[i] = (_Float16)((xp[i] - mean) * rs * sc[i] + bs[i]);
}

// ---------------------------------------------------------------------------
// Fused-epilogue WMMA GEMM:  out = act(A_f16 @ W_f32 + bias) (+ resid)
// A: MxK f16 row-major; W: KxN f32 row-major. BM=BN=64, BK=32, 8 waves.
// act: 0 none, 1 silu, 2 gelu(tanh), 3 log-sigmoid
// ---------------------------------------------------------------------------
DEV float apply_act(float v, int act) {
  if (act == 1) return v / (1.f + __expf(-v));
  if (act == 2) {
    float c = 0.7978845608028654f * (v + 0.044715f * v * v * v);
    return 0.5f * v * (1.f + tanhf(c));
  }
  if (act == 3) return fminf(v, 0.f) - log1pf(__expf(-fabsf(v)));
  return v;
}

__global__ __launch_bounds__(256) void gemm_wmma_kernel(
    const _Float16* __restrict__ A, const float* __restrict__ W,
    const float* __restrict__ bias, const float* __restrict__ resid,
    float* __restrict__ out32, _Float16* __restrict__ out16,
    int M, int N, int K, int act) {
  __shared__ _Float16 Ash[64][32];   // row stride 64B (16B aligned chunks)
  __shared__ _Float16 Bsh[32][64];   // row = K, 128B stride (32B aligned runs)

  int tid = threadIdx.x;
  int lane = tid & 31, wv = tid >> 5;
  int rt = wv & 3;                 // row tile (16 rows)
  int ctBase = (wv >> 2) * 2;      // two col tiles per wave
  int gm0 = blockIdx.y * 64;
  int gn0 = blockIdx.x * 64;

  int arow = tid >> 2;             // 0..63
  int acol = (tid & 3) * 8;        // 0,8,16,24 (16B chunks)
  int brow = tid & 31;             // 0..31
  int bcol0 = (tid >> 5) * 8;      // 0..56

  bool fullN = (gn0 + 64 <= N);    // uniform fast/slow path select

  v8f acc0 = {}, acc1 = {};
  int half = lane >> 4, nn = lane & 15;

  for (int kk = 0; kk < K; kk += 32) {
    // ---- A tile: async 16B copy per thread ----
    const _Float16* ga = A + (size_t)(gm0 + arow) * K + kk + acol;
#if USE_ASYNC_COPY
    __builtin_amdgcn_global_load_async_to_lds_b128(
        (__attribute__((address_space(1))) v4i*)ga,
        (__attribute__((address_space(3))) v4i*)&Ash[arow][acol], 0, 0);
#else
    *(H8*)&Ash[arow][acol] = *(const H8*)ga;
#endif
    // ---- B tile: f32 -> f16, vectorized fast path ----
    const float* gw = W + (size_t)(kk + brow) * N + gn0;
    if (fullN) {
      float4 w0 = *(const float4*)(gw + bcol0);
      float4 w1 = *(const float4*)(gw + bcol0 + 4);
      H8 hb;
      hb.v[0] = (_Float16)w0.x; hb.v[1] = (_Float16)w0.y;
      hb.v[2] = (_Float16)w0.z; hb.v[3] = (_Float16)w0.w;
      hb.v[4] = (_Float16)w1.x; hb.v[5] = (_Float16)w1.y;
      hb.v[6] = (_Float16)w1.z; hb.v[7] = (_Float16)w1.w;
      *(H8*)&Bsh[brow][bcol0] = hb;
    } else {
      const float* wrow = W + (size_t)(kk + brow) * N;
#pragma unroll
      for (int i = 0; i < 8; ++i) {
        int c = bcol0 + i;
        int gc = gn0 + c;
        int cc = gc < N ? gc : N - 1;       // clamped, always-valid load
        float v = wrow[cc];
        if (gc >= N) v = 0.f;
        Bsh[brow][c] = (_Float16)v;
      }
    }
#if USE_ASYNC_COPY
    async_wait0();
#endif
    __syncthreads();

    // A fragment: two contiguous 16B chunks of row (16*rt+nn)
    const v8h* arp = (const v8h*)&Ash[16 * rt + nn][0];
    v16h af = cat8(arp[half], arp[2 + half]);
    // B fragments: lane%16 = N column, element s = K row (table order)
    v16h bf0, bf1;
#pragma unroll
    for (int s = 0; s < 16; ++s) {
      int kr = (s < 8) ? (8 * half + s) : (16 + 8 * half + (s - 8));
      bf0[s] = Bsh[kr][16 * ctBase + nn];
      bf1[s] = Bsh[kr][16 * (ctBase + 1) + nn];
    }

    acc0 = wmma_f16(af, bf0, acc0);
    acc1 = wmma_f16(af, bf1, acc1);
    __syncthreads();
  }

#pragma unroll
  for (int t2 = 0; t2 < 2; ++t2) {
    v8f acc = t2 ? acc1 : acc0;
    int gcol = gn0 + 16 * (ctBase + t2) + nn;
    if (gcol < N) {
#pragma unroll
      for (int j = 0; j < 8; ++j) {
        int grow = gm0 + 16 * rt + j + 8 * half;
        float v = acc[j];
        if (bias) v += bias[gcol];
        v = apply_act(v, act);
        size_t idx = (size_t)grow * N + gcol;
        if (resid) v += resid[idx];
        if (out32) out32[idx] = v;
        if (out16) out16[idx] = (_Float16)v;
      }
    }
  }
}

// ---------------------------------------------------------------------------
// Time-decay retention attention (one wave per (b,h,16-query tile)).
// score(n,m) = (q_n . k_m)/8 * exp(g_n*(t_n-t_m)) for m<=n  (exponent <= 0)
// out(n)    = sum_m score*v_m / max(1, sum_m |score|)
// Key blocks of 32: QK^T = 4 WMMAs, PV = 4 WMMAs per block.
// V fragments load straight from global (row-major = native B layout);
// K^T needs an LDS transpose stage.
// ---------------------------------------------------------------------------
__global__ __launch_bounds__(32) void attn_kernel(const _Float16* __restrict__ qkvh,
                                                  const float* __restrict__ itime,
                                                  const float* __restrict__ logg,
                                                  float* __restrict__ attnout) {
  int qt = blockIdx.x, hh = blockIdx.y, bb = blockIdx.z;
  int lane = threadIdx.x;
  int nn = lane & 15, half = lane >> 4;
  int n0 = qt * 16;

  __shared__ float tq[16], gq[16], tk[32], norm16[16];
  __shared__ _Float16 P[16][32];    // post-decay scores, A-layout source
  __shared__ _Float16 Kt[64][32];   // K^T: row = head dim, col = key

  if (lane < 16) {
    tq[lane] = itime[bb * T + n0 + lane];
    gq[lane] = logg[(size_t)(bb * T + n0 + lane) * H + hh];
  }
  __syncthreads();

  // Q fragments (A layout): pure vector loads; 1/sqrt(DK) folded into decay
  const v8h* qr = (const v8h*)(qkvh + (size_t)(bb * T + n0 + nn) * QKV_STRIDE + hh * 64);
  v16h aq0 = cat8(qr[half], qr[2 + half]);       // dims 0..31
  v16h aq1 = cat8(qr[4 + half], qr[6 + half]);   // dims 32..63

  v8f oacc[4]; oacc[0] = {}; oacc[1] = {}; oacc[2] = {}; oacc[3] = {};
  float np[8];
#pragma unroll
  for (int j = 0; j < 8; ++j) np[j] = 0.f;

  int nkb = (n0 + 15) / 32 + 1;
  for (int mb = 0; mb < nkb; ++mb) {
    int base = mb * 32;
    tk[lane] = itime[bb * T + base + lane];
    // stage K block transposed: lane = key row, scatter into Kt[dim][key]
    {
      const _Float16* kr = qkvh + (size_t)(bb * T + base + lane) * QKV_STRIDE + QKD + hh * 64;
      v8h kv[8];
#pragma unroll
      for (int c = 0; c < 8; ++c) kv[c] = *(const v8h*)(kr + 8 * c);
#pragma unroll
      for (int c = 0; c < 8; ++c)
#pragma unroll
        for (int e = 0; e < 8; ++e)
          Kt[8 * c + e][lane] = kv[c][e];
    }
    __syncthreads();

#pragma unroll
    for (int nt = 0; nt < 2; ++nt) {
      // B fragments for Q@K^T: lane%16 = key, element s = contraction dim (table order)
      v16h bk0, bk1;
#pragma unroll
      for (int s = 0; s < 16; ++s) {
        int kx = (s < 8) ? (8 * half + s) : (16 + 8 * half + (s - 8));
        bk0[s] = Kt[kx][16 * nt + nn];        // dims 0..31
        bk1[s] = Kt[32 + kx][16 * nt + nn];   // dims 32..63
      }
      v8f s = {};
      s = wmma_f16(aq0, bk0, s);
      s = wmma_f16(aq1, bk1, s);

      int keyi = base + 16 * nt + nn;
      float tkv = tk[16 * nt + nn];
#pragma unroll
      for (int j = 0; j < 8; ++j) {
        int rloc = j + 8 * half;
        int rowg = n0 + rloc;
        float p = 0.f;
        if (keyi <= rowg)
          p = s[j] * (0.125f * __expf(gq[rloc] * (tq[rloc] - tkv)));
        P[rloc][16 * nt + nn] = (_Float16)p;
        np[j] += fabsf(p);
      }
    }
    __syncthreads();

    // P (16x32, A layout: two contiguous chunks per lane) @ V block (32x64)
    const v8h* pr = (const v8h*)&P[nn][0];
    v16h ap = cat8(pr[half], pr[2 + half]);
#pragma unroll
    for (int nt2 = 0; nt2 < 4; ++nt2) {
      // V B-fragment directly from global: lane%16 = dim, element s = key (table order)
      v16h bv;
#pragma unroll
      for (int s = 0; s < 16; ++s) {
        int kx = (s < 8) ? (8 * half + s) : (16 + 8 * half + (s - 8));
        bv[s] = qkvh[(size_t)(bb * T + base + kx) * QKV_STRIDE +
                     2 * QKD + hh * 64 + 16 * nt2 + nn];
      }
      oacc[nt2] = wmma_f16(ap, bv, oacc[nt2]);
    }
    __syncthreads();
  }

  // row-wise |score| sums: reduce within each 16-lane half
#pragma unroll
  for (int j = 0; j < 8; ++j) {
    float v = np[j];
    v += __shfl_xor(v, 1, 32);
    v += __shfl_xor(v, 2, 32);
    v += __shfl_xor(v, 4, 32);
    v += __shfl_xor(v, 8, 32);
    np[j] = v;
  }
  if (nn == 0) {
#pragma unroll
    for (int j = 0; j < 8; ++j) norm16[j + 8 * half] = np[j];
  }
  __syncthreads();

#pragma unroll
  for (int nt2 = 0; nt2 < 4; ++nt2) {
#pragma unroll
    for (int j = 0; j < 8; ++j) {
      int rloc = j + 8 * half;
      float v = oacc[nt2][j] / fmaxf(norm16[rloc], 1.f);
      attnout[(size_t)(bb * T + n0 + rloc) * VDIM + hh * 64 + 16 * nt2 + nn] = v;
    }
  }
}

// ---------------------------------------------------------------------------
// gate:  goh = silu(x1@Wg) * attn_out   (silu already applied in GEMM epilogue)
// ---------------------------------------------------------------------------
__global__ __launch_bounds__(256) void gate_kernel(const float* __restrict__ g,
                                                   const float* __restrict__ a,
                                                   _Float16* __restrict__ o) {
  int i = blockIdx.x * 256 + threadIdx.x;
  o[i] = (_Float16)(g[i] * a[i]);
}

// ---------------------------------------------------------------------------
static void gemm_launch(hipStream_t s, const _Float16* A, const float* W,
                        const float* bias, const float* resid,
                        float* o32, _Float16* o16, int M, int N, int K, int act) {
  dim3 grid((N + 63) / 64, (M + 63) / 64);
  gemm_wmma_kernel<<<grid, 256, 0, s>>>(A, W, bias, resid, o32, o16, M, N, K, act);
}

extern "C" void kernel_launch(void* const* d_in, const int* in_sizes, int n_in,
                              void* d_out, int out_size, void* d_ws, size_t ws_size,
                              hipStream_t stream) {
  (void)in_sizes; (void)n_in; (void)out_size; (void)ws_size;
  const int*   X     = (const int*)  d_in[0];
  const float* itime = (const float*)d_in[1];
  const float* embed = (const float*)d_in[2];
  const float* sos   = (const float*)d_in[3];
  const float* ln1_s = (const float*)d_in[4];
  const float* ln1_b = (const float*)d_in[5];
  const float* Wqkv  = (const float*)d_in[6];
  const float* bqkv  = (const float*)d_in[7];
  const float* Wdec  = (const float*)d_in[8];
  const float* Wg    = (const float*)d_in[9];
  const float* Wo    = (const float*)d_in[10];
  const float* bo    = (const float*)d_in[11];
  const float* ln2_s = (const float*)d_in[12];
  const float* ln2_b = (const float*)d_in[13];
  const float* Wff1  = (const float*)d_in[14];
  const float* bff1  = (const float*)d_in[15];
  const float* Wff2  = (const float*)d_in[16];
  const float* bff2  = (const float*)d_in[17];
  const float* lnf_s = (const float*)d_in[18];
  const float* lnf_b = (const float*)d_in[19];
  const float* Whead = (const float*)d_in[20];
  const float* bhead = (const float*)d_in[21];

  char* ws = (char*)d_ws;
  size_t off = 0;
  auto alloc = [&](size_t bytes) {
    void* p = ws + off;
    off = (off + bytes + 255) & ~(size_t)255;
    return p;
  };
  float*    hbuf  = (float*)   alloc((size_t)M_TOK * D * 4);
  _Float16* xh    = (_Float16*)alloc((size_t)M_TOK * D * 2);
  _Float16* qkvh  = (_Float16*)alloc((size_t)M_TOK * QKV_N * 2);
  float*    logg  = (float*)   alloc((size_t)M_TOK * H * 4);
  float*    gbuf  = (float*)   alloc((size_t)M_TOK * VDIM * 4);
  float*    attno = (float*)   alloc((size_t)M_TOK * VDIM * 4);
  _Float16* goh   = (_Float16*)alloc((size_t)M_TOK * VDIM * 2);
  _Float16* ffh   = (_Float16*)alloc((size_t)M_TOK * FDIM * 2);

  embed_kernel<<<M_TOK, 256, 0, stream>>>(X, embed, sos, hbuf);

  for (int l = 0; l < LAYERS; ++l) {
    ln_kernel<<<M_TOK, 256, 0, stream>>>(hbuf, ln1_s + l * D, ln1_b + l * D, xh);
    gemm_launch(stream, xh, Wqkv + (size_t)l * D * QKV_N, bqkv + (size_t)l * QKV_N,
                nullptr, nullptr, qkvh, M_TOK, QKV_N, D, 0);
    gemm_launch(stream, xh, Wdec + (size_t)l * D * H, nullptr,
                nullptr, logg, nullptr, M_TOK, H, D, 3);
    gemm_launch(stream, xh, Wg + (size_t)l * D * VDIM, nullptr,
                nullptr, gbuf, nullptr, M_TOK, VDIM, D, 1);
    attn_kernel<<<dim3(T / 16, H, B), 32, 0, stream>>>(qkvh, itime, logg, attno);
    gate_kernel<<<(M_TOK * VDIM) / 256, 256, 0, stream>>>(gbuf, attno, goh);
    gemm_launch(stream, goh, Wo + (size_t)l * VDIM * D, bo + (size_t)l * D,
                hbuf, hbuf, nullptr, M_TOK, D, VDIM, 0);
    ln_kernel<<<M_TOK, 256, 0, stream>>>(hbuf, ln2_s + l * D, ln2_b + l * D, xh);
    gemm_launch(stream, xh, Wff1 + (size_t)l * D * FDIM, bff1 + (size_t)l * FDIM,
                nullptr, nullptr, ffh, M_TOK, FDIM, D, 2);
    gemm_launch(stream, ffh, Wff2 + (size_t)l * FDIM * D, bff2 + (size_t)l * D,
                hbuf, hbuf, nullptr, M_TOK, D, FDIM, 0);
  }

  ln_kernel<<<M_TOK, 256, 0, stream>>>(hbuf, lnf_s, lnf_b, xh);
  gemm_launch(stream, xh, Whead, bhead, nullptr, (float*)d_out, nullptr,
              M_TOK, VOCAB, D, 0);
}
